// SpGAT_41996190220663
// MI455X (gfx1250) — compile-verified, matrix operand order checked by reading, not run on hardware
//
#include <hip/hip_runtime.h>
#include <hip/hip_bf16.h>
#include <math.h>

typedef __attribute__((ext_vector_type(16))) _Float16 v16h;
typedef __attribute__((ext_vector_type(8)))  _Float16 h8v;
typedef __attribute__((ext_vector_type(8)))  float    v8f;

static __device__ __forceinline__ v16h cat8(h8v lo, h8v hi) {
  return __builtin_shufflevector(lo, hi, 0,1,2,3,4,5,6,7,8,9,10,11,12,13,14,15);
}

static __device__ __forceinline__ v8f wmma_f16(v16h a, v16h b, v8f c) {
  // D = A(16x32 f16) * B(32x16 f16) + C(16x16 f32)
  return __builtin_amdgcn_wmma_f32_16x16x32_f16(false, a, false, b, (short)0, c, false, false);
}

// ---------------------------------------------------------------------------
// Pack adj int32[8192][8192] -> bitmask u32[8192][256]. One output word/thread.
// ---------------------------------------------------------------------------
__global__ void k_pack_adj(const int* __restrict__ adj, unsigned* __restrict__ bits) {
  unsigned t = blockIdx.x * blockDim.x + threadIdx.x;      // word index
  const int4* p = (const int4*)adj + (size_t)t * 8;
  unsigned m = 0;
#pragma unroll
  for (int q = 0; q < 8; ++q) {
    int4 v = p[q];
    m |= (v.x > 0 ? 1u : 0u) << (q * 4 + 0);
    m |= (v.y > 0 ? 1u : 0u) << (q * 4 + 1);
    m |= (v.z > 0 ? 1u : 0u) << (q * 4 + 2);
    m |= (v.w > 0 ? 1u : 0u) << (q * 4 + 3);
  }
  bits[t] = m;
}

__global__ void k_f32_to_f16(const float* __restrict__ src, _Float16* __restrict__ dst) {
  size_t t = (size_t)blockIdx.x * blockDim.x + threadIdx.x;
  dst[t] = (_Float16)src[t];
}

// W[K][Nn] f32 -> Wt[Nn][K] f16 (per-head via blockIdx.y)
__global__ void k_wtrans(const float* __restrict__ W, _Float16* __restrict__ Wt, int K, int Nn) {
  size_t hoff = (size_t)blockIdx.y * K * Nn;
  int t = blockIdx.x * blockDim.x + threadIdx.x;
  int k = t / Nn, c = t % Nn;
  Wt[hoff + (size_t)c * K + k] = (_Float16)W[hoff + t];
}

// ---------------------------------------------------------------------------
// C^T = (A @ B)^T :  A[8192][K] f16 row-major, Bt[N][K] f16, Ct[N][8192] f16.
// Block = 8 waves * 16 rows = 128 rows; each wave owns NT 16x16 tiles (N=NT*16).
// ---------------------------------------------------------------------------
template <int NT>
__global__ __launch_bounds__(256) void k_gemm_tn(const _Float16* __restrict__ A,
                                                 const _Float16* __restrict__ Bt,
                                                 _Float16* __restrict__ Ct,
                                                 int K, int bHeadStride, int cHeadStride) {
  Bt += (size_t)blockIdx.y * bHeadStride;
  Ct += (size_t)blockIdx.y * cHeadStride;
  const int lane = threadIdx.x & 31;
  const int wv   = threadIdx.x >> 5;
  const int half = lane >> 4;
  const int l16  = lane & 15;
  const int rowb = blockIdx.x * 128 + wv * 16;
  const size_t arow = (size_t)(rowb + l16) * K;

  v8f acc[NT];
#pragma unroll
  for (int t = 0; t < NT; ++t) acc[t] = {};

  for (int kc = 0; kc < K; kc += 32) {
    const int kb = kc + 8 * half;                       // CDNA5 16-bit A/B layout
    v16h a = cat8(*(const h8v*)(A + arow + kb),
                  *(const h8v*)(A + arow + kb + 16));
#pragma unroll
    for (int t = 0; t < NT; ++t) {
      const _Float16* bp = Bt + (size_t)(t * 16 + l16) * K + kb;
      v16h b = cat8(*(const h8v*)bp, *(const h8v*)(bp + 16));
      acc[t] = wmma_f16(a, b, acc[t]);
    }
  }
  // C layout: VGPR g -> row rowb + 8*half + g, col = tile*16 + l16. Store transposed.
#pragma unroll
  for (int t = 0; t < NT; ++t) {
    h8v o;
#pragma unroll
    for (int g = 0; g < 8; ++g) o[g] = (_Float16)acc[t][g];
    *(h8v*)(Ct + (size_t)(t * 16 + l16) * 8192 + rowb + 8 * half) = o;
  }
}

// s1[n] = sum_j WhT[j][n]*a[j], s2[n] = sum_j WhT[j][n]*a[F+j]
__global__ void k_s1s2(const _Float16* __restrict__ WhT, const float* __restrict__ a,
                       float* __restrict__ s1, float* __restrict__ s2,
                       int F, int whHeadStride, int aHeadStride) {
  const int h = blockIdx.y;
  WhT += (size_t)h * whHeadStride;
  a   += (size_t)h * aHeadStride;
  s1  += (size_t)h * 8192;
  s2  += (size_t)h * 8192;
  int n = blockIdx.x * blockDim.x + threadIdx.x;
  float v1 = 0.f, v2 = 0.f;
  for (int j = 0; j < F; ++j) {
    float w = (float)WhT[(size_t)j * 8192 + n];
    v1 += w * a[j];
    v2 += w * a[F + j];
  }
  s1[n] = v1;
  s2[n] = v2;
}

// ---------------------------------------------------------------------------
// Single-pass fused attention: O = (exp_mask @ Wh) / rowsum.
// e = lrelu(s1[r]+s2[c]) is bounded, so unnormalized exp(e) is f16-safe and no
// separate max/sum pass is needed; rowsum is accumulated in-register next to
// the WMMA and merged with one shfl_xor(16) (the two lane-halves of a row).
// B (WhT) is staged through double-buffered LDS using CDNA5 async global->LDS
// loads (ASYNCcnt): the next 128-K stage streams into LDS while the current
// stage's exp/WMMA work executes; one s_wait_asynccnt + barrier per stage.
// FINAL=false: write f16 into hcat at column offset head*64.
// FINAL=true : apply ELU, write f32 to out[8192][16].
// ---------------------------------------------------------------------------
template <int NT, bool FINAL>
__global__ __launch_bounds__(256) void k_attn(const _Float16* __restrict__ WhT,
                                              const unsigned* __restrict__ bits,
                                              const float* __restrict__ s1,
                                              const float* __restrict__ s2,
                                              int whHeadStride,
                                              _Float16* __restrict__ hcat,
                                              float* __restrict__ outf) {
  constexpr int KSTAGE = 128;
  constexpr int KPAD   = KSTAGE + 8;        // pad to spread LDS banks
  constexpr int NSTAGE = 8192 / KSTAGE;     // 64
  constexpr int NCOLS  = NT * 16;
  __shared__ _Float16 lb[2][NCOLS * KPAD];

  const int h = blockIdx.y;
  WhT += (size_t)h * whHeadStride;
  s1  += (size_t)h * 8192;
  s2  += (size_t)h * 8192;
  const int tid  = threadIdx.x;
  const int lane = tid & 31;
  const int wv   = tid >> 5;
  const int half = lane >> 4;
  const int l16  = lane & 15;
  const int rowb = blockIdx.x * 128 + wv * 16;
  const int arow = rowb + l16;              // A-fragment row for this lane
  const float s1r = s1[arow];
  const unsigned* brow = bits + (size_t)arow * 256;

  v8f acc[NT];
#pragma unroll
  for (int t = 0; t < NT; ++t) acc[t] = {};
  float psum = 0.f;

  // async stage fill: NT 16B pieces per thread, coalesced along K, no VGPR data
  auto fill_async = [&](int ks, int bufIdx) {
#pragma unroll
    for (int j = 0; j < NT; ++j) {
      int p = tid + j * 256, col = p >> 4, k0 = (p & 15) * 8;
      unsigned lofs = (unsigned)(uintptr_t)(&lb[bufIdx][col * KPAD + k0]);
      unsigned long long g =
          (unsigned long long)(uintptr_t)(WhT + (size_t)col * 8192 + ks + k0);
      asm volatile("global_load_async_to_lds_b128 %0, %1, off"
                   :: "v"(lofs), "v"(g) : "memory");
    }
  };

  fill_async(0, 0);
  asm volatile("s_wait_asynccnt 0x0" ::: "memory");
  __syncthreads();

  for (int s = 0; s < NSTAGE; ++s) {
    const int cur = s & 1;
    if (s + 1 < NSTAGE)                      // fire-and-forget prefetch of next stage
      fill_async((s + 1) * KSTAGE, cur ^ 1);

    const int ksb = s * KSTAGE;
#pragma unroll
    for (int kk = 0; kk < KSTAGE; kk += 32) {
      const int kc = ksb + kk;
      const int kb = kc + 8 * half;
      const unsigned aw = brow[kc >> 5];
      float4 sa = *(const float4*)(s2 + kb);
      float4 sb = *(const float4*)(s2 + kb + 4);
      float4 sc = *(const float4*)(s2 + kb + 16);
      float4 sd = *(const float4*)(s2 + kb + 20);
      float lo[8] = {sa.x, sa.y, sa.z, sa.w, sb.x, sb.y, sb.z, sb.w};
      float hi[8] = {sc.x, sc.y, sc.z, sc.w, sd.x, sd.y, sd.z, sd.w};
      v16h a;
#pragma unroll
      for (int j = 0; j < 8; ++j) {
        float z  = s1r + lo[j];
        float e  = z  > 0.f ? z  : 0.2f * z;             // leaky_relu(0.2)
        float p  = ((aw >> (8 * half + j)) & 1u)      ? __expf(e)  : 0.f;
        float z2 = s1r + hi[j];
        float e2 = z2 > 0.f ? z2 : 0.2f * z2;
        float p2 = ((aw >> (8 * half + 16 + j)) & 1u) ? __expf(e2) : 0.f;
        psum += p + p2;
        a[j]     = (_Float16)p;
        a[8 + j] = (_Float16)p2;
      }
      const int lofs = kk + 8 * half;
#pragma unroll
      for (int t = 0; t < NT; ++t) {
        const _Float16* bp = &lb[cur][(t * 16 + l16) * KPAD + lofs];
        v16h b = cat8(*(const h8v*)bp, *(const h8v*)(bp + 16));
        acc[t] = wmma_f16(a, b, acc[t]);
      }
    }
    if (s + 1 < NSTAGE)
      asm volatile("s_wait_asynccnt 0x0" ::: "memory");
    __syncthreads();
  }

  // rowsum for row rowb+l16: this lane's half-K plus the mirror lane's half
  float rs = psum + __shfl_xor(psum, 16, 32);
  float rinv = rs > 0.f ? 1.f / rs : 0.f;

  if constexpr (FINAL) {
#pragma unroll
    for (int g = 0; g < 8; ++g) {
      float rg = __shfl(rinv, 8 * half + g, 32);         // rinv of row rowb+8*half+g
      float v = acc[0][g] * rg;
      float o = v > 0.f ? v : __expf(v) - 1.f;           // ELU
      outf[(size_t)(rowb + 8 * half + g) * 16 + l16] = o;
    }
  } else {
#pragma unroll
    for (int g = 0; g < 8; ++g) {
      float rg = __shfl(rinv, 8 * half + g, 32);
#pragma unroll
      for (int t = 0; t < NT; ++t)
        hcat[(size_t)(rowb + 8 * half + g) * 512 + h * 64 + t * 16 + l16] =
            (_Float16)(acc[t][g] * rg);
    }
  }
}

extern "C" void kernel_launch(void* const* d_in, const int* in_sizes, int n_in,
                              void* d_out, int out_size, void* d_ws, size_t ws_size,
                              hipStream_t stream) {
  (void)in_sizes; (void)n_in; (void)out_size; (void)ws_size;
  const float* x       = (const float*)d_in[0];
  const int*   adj     = (const int*)d_in[1];
  const float* W_heads = (const float*)d_in[2];
  const float* a_heads = (const float*)d_in[3];
  const float* W_out   = (const float*)d_in[4];
  const float* a_out   = (const float*)d_in[5];
  float* out = (float*)d_out;

  constexpr int N = 8192, FIN = 512, FH = 64, H = 8, NC = 16;
  char* ws = (char*)d_ws;
  unsigned*  adjbits = (unsigned*)(ws);                       //  0MB: 8 MB bitmask
  _Float16*  xh      = (_Float16*)(ws + (8ull  << 20));       //  8MB: x in f16 (8 MB)
  _Float16*  WtAll   = (_Float16*)(ws + (16ull << 20));       // 16MB: W_heads^T f16 (512 KB)
  _Float16*  WoutT   = (_Float16*)(ws + (17ull << 20));       // 17MB: W_out^T f16 (16 KB)
  _Float16*  WhT     = (_Float16*)(ws + (18ull << 20));       // 18MB: Wh^T per head (8 MB)
  float*     s1      = (float*)  (ws + (26ull << 20));        // 26MB: [9][8192] stats
  float*     s2      = s1 + 9 * N;
  _Float16*  hcat    = (_Float16*)(ws + (28ull << 20));       // 28MB: concat h f16 (8 MB)
  _Float16*  Wh2T    = (_Float16*)(ws + (36ull << 20));       // 36MB: layer2 Wh^T (256 KB)

  // ---- prep: bitmask + f16 conversions ----
  k_pack_adj  <<<dim3((N / 32) * N / 256), 256, 0, stream>>>(adj, adjbits);
  k_f32_to_f16<<<dim3(N * FIN / 256),      256, 0, stream>>>(x, xh);
  k_wtrans    <<<dim3(FIN * FH / 256, H),  256, 0, stream>>>(W_heads, WtAll, FIN, FH);
  k_wtrans    <<<dim3(FIN * NC / 256, 1),  256, 0, stream>>>(W_out, WoutT, FIN, NC);

  // ---- layer 1 (8 heads) ----
  k_gemm_tn<4><<<dim3(N / 128, H), 256, 0, stream>>>(xh, WtAll, WhT, FIN, FIN * FH, FH * N);
  k_s1s2      <<<dim3(N / 256, H), 256, 0, stream>>>(WhT, a_heads, s1, s2, FH, FH * N, 2 * FH);
  k_attn<4, false><<<dim3(N / 128, H), 256, 0, stream>>>(WhT, adjbits, s1, s2,
                                                         FH * N, hcat, nullptr);

  // ---- layer 2 (output) ----
  k_gemm_tn<1><<<dim3(N / 128, 1), 256, 0, stream>>>(hcat, WoutT, Wh2T, FIN, 0, 0);
  k_s1s2      <<<dim3(N / 256, 1), 256, 0, stream>>>(Wh2T, a_out, s1 + 8 * N, s2 + 8 * N, NC, 0, 0);
  k_attn<1, true><<<dim3(N / 128, 1), 256, 0, stream>>>(Wh2T, adjbits, s1 + 8 * N, s2 + 8 * N,
                                                        0, nullptr, out);
}
